// GraphNet_29343216566653
// MI455X (gfx1250) — matrix operand order repaired and checked by reading, not compiled, not run on hardware
//
#include <hip/hip_runtime.h>
#include <stdint.h>

#define N_NODES  50000
#define N_EDGES  1600000
#define NODE_F   64
#define EDGE_F   32
#define HID      128
#define EDGE_OUT 64
#define EDGE_IN  160   // 2*NODE_F + EDGE_F

// ---- workspace byte offsets (all 16B aligned) ----
#define WS_AGG   0              // float  [N_NODES][64]   = 12,800,000 B
#define WS_XBF   12800000       // bf16   [N_NODES][64]   =  6,400,000 B
#define WS_WT1   19200000       // bf16   [128][160] (W1^T) = 40,960 B
#define WS_WT2   19240960       // bf16   [64][128]  (W2^T) = 16,384 B
#define WS_WN1T  19257344       // bf16   [128][128] (Wn1^T)= 32,768 B
#define WS_WN2T  19290112       // bf16   [64][128]  (Wn2^T)= 16,384 B

// ---- edge-kernel dynamic LDS layout (bytes) ----
#define EOFF_IDX 0              // int row[128], col[128] = 1024
#define EOFF_A   1024           // bf16 [128][168]  A-tile, reused as H-tile
#define EOFF_W1  44032          // bf16 [128][168]  W1^T
#define EOFF_W2  87040          // bf16 [64][136]   W2^T
#define EOFF_B1  104448         // float[128]
#define EOFF_B2  104960         // float[64]
#define ESMEM    105216

// ---- node-kernel dynamic LDS layout (bytes) ----
#define NOFF_A   0              // bf16 [128][136]  A-tile, reused as H-tile
#define NOFF_W1  34816          // bf16 [128][136]  Wn1^T
#define NOFF_W2  69632          // bf16 [64][136]   Wn2^T
#define NOFF_B1  87040          // float[128]
#define NOFF_B2  87552          // float[64]
#define NSMEM    87808

typedef __attribute__((ext_vector_type(16))) __bf16 v16bf;
typedef __attribute__((ext_vector_type(8)))  float  v8f;

union Frag { v16bf bf; uint4 q[2]; };

// f32 -> bf16 (RNE); prefer v_cvt_pk_bf16_f32 when the builtin exists
__device__ __forceinline__ unsigned int pk2bf(float lo, float hi) {
#if __has_builtin(__builtin_amdgcn_cvt_pk_bf16_f32)
  typedef __attribute__((ext_vector_type(2))) __bf16 v2bf;
  v2bf r = __builtin_amdgcn_cvt_pk_bf16_f32(lo, hi);
  return __builtin_bit_cast(unsigned int, r);
#else
  unsigned int a = __builtin_bit_cast(unsigned int, lo);
  unsigned int b = __builtin_bit_cast(unsigned int, hi);
  a += 0x7FFFu + ((a >> 16) & 1u);
  b += 0x7FFFu + ((b >> 16) & 1u);
  return (a >> 16) | (b & 0xFFFF0000u);
#endif
}
__device__ __forceinline__ unsigned short f2bf(float f) {
  return (unsigned short)pk2bf(f, 0.f);   // low half
}
__device__ __forceinline__ void atomicAddF(float* p, float v) {
  __hip_atomic_fetch_add(p, v, __ATOMIC_RELAXED, __HIP_MEMORY_SCOPE_AGENT);
}

// ---------------- prep: zero aggregation buffer ----------------
__global__ void __launch_bounds__(256) zero_agg(float4* agg4) {
  int i = blockIdx.x * 256 + threadIdx.x;   // 800,000 float4 = 3,200,000 floats
  agg4[i] = float4{0.f, 0.f, 0.f, 0.f};
}

// ---------------- prep: f32 -> bf16 conversions (+ weight transpose) --------
__global__ void __launch_bounds__(256) prep_convert(
    const float* __restrict__ x,
    const float* __restrict__ We1, const float* __restrict__ We2,
    const float* __restrict__ Wn1, const float* __restrict__ Wn2,
    unsigned int* __restrict__ xbf, unsigned int* __restrict__ wt1,
    unsigned int* __restrict__ wt2, unsigned int* __restrict__ wn1t,
    unsigned int* __restrict__ wn2t)
{
  const int XT  = N_NODES * 32;   // x: packed dwords
  const int W1T = 128 * 80;       // W1^T dwords
  const int W2T = 64 * 64;
  const int N1T = 128 * 64;
  const int N2T = 64 * 64;
  const int total = XT + W1T + W2T + N1T + N2T;
  const float2* x2 = (const float2*)x;
  for (int i = blockIdx.x * 256 + threadIdx.x; i < total; i += gridDim.x * 256) {
    if (i < XT) {
      float2 p = x2[i];
      xbf[i] = pk2bf(p.x, p.y);
    } else if (i < XT + W1T) {
      int t = i - XT; int n = t / 80; int k = (t % 80) * 2;
      wt1[n * 80 + (k >> 1)] = pk2bf(We1[k * 128 + n], We1[(k + 1) * 128 + n]);
    } else if (i < XT + W1T + W2T) {
      int t = i - XT - W1T; int n = t / 64; int k = (t % 64) * 2;
      wt2[n * 64 + (k >> 1)] = pk2bf(We2[k * 64 + n], We2[(k + 1) * 64 + n]);
    } else if (i < XT + W1T + W2T + N1T) {
      int t = i - XT - W1T - W2T; int n = t / 64; int k = (t % 64) * 2;
      wn1t[n * 64 + (k >> 1)] = pk2bf(Wn1[k * 128 + n], Wn1[(k + 1) * 128 + n]);
    } else {
      int t = i - XT - W1T - W2T - N1T; int n = t / 64; int k = (t % 64) * 2;
      wn2t[n * 64 + (k >> 1)] = pk2bf(Wn2[k * 64 + n], Wn2[(k + 1) * 64 + n]);
    }
  }
}

// ---- WMMA fragment loaders (layouts per CDNA5 ISA §7.12.2) ----
// A 16x32 bf16, lane (r=L&15, kh=L>>4): regs 0-3 = K[k0+8kh .. +8) (16B),
//   regs 4-7 = K[k0+16+8kh .. +8) (16B)  -> two ds_load_b128.
// B 32x16 bf16 from W^T[n][k]: lane col n=r holds K[k0+8kh .. +8) and
//   K[k0+16+8kh .. +8) (same order as A) -> two ds_load_b128.
__device__ __forceinline__ v16bf loadA(const uint4* row4, int k0, int kh) {
  Frag f;
  f.q[0] = row4[(k0 >> 3) + kh];
  f.q[1] = row4[(k0 >> 3) + 2 + kh];
  return f.bf;
}
__device__ __forceinline__ v16bf loadB(const uint4* col4, int k0, int kh) {
  Frag f;
  f.q[0] = col4[(k0 >> 3) + kh];
  f.q[1] = col4[(k0 >> 3) + 2 + kh];
  return f.bf;
}

// ---------------- edge MLP + scatter-add ----------------
__global__ void __launch_bounds__(256) edge_kernel(
    const int*   __restrict__ eidx,    // [2][E]
    const float* __restrict__ eattr,   // [E][32]
    const float* __restrict__ be1, const float* __restrict__ be2,
    const unsigned int* __restrict__ xbf,  // [N][32] dwords (bf16 pairs)
    const unsigned int* __restrict__ wt1,  // [128][80] dwords
    const unsigned int* __restrict__ wt2,  // [64][64] dwords
    float* __restrict__ agg,               // [N][64]
    float* __restrict__ eout)              // [E][64]
{
  extern __shared__ char smem[];
  int* sRow = (int*)(smem + EOFF_IDX);
  float* sB1 = (float*)(smem + EOFF_B1);
  float* sB2 = (float*)(smem + EOFF_B2);

  const int tid  = threadIdx.x;
  const int base = blockIdx.x * 128;     // E = 12500 * 128 exactly

  if (tid < 128) {
    sRow[tid] = eidx[base + tid];
    sB1[tid]  = be1[tid];
    if (tid < 64) sB2[tid] = be2[tid];
  }
  { // stage W1^T: 128 rows x 80 dwords -> LDS stride 84 dwords
    int rw = tid >> 1, hf = tid & 1;
    const uint4* src = (const uint4*)(wt1 + rw * 80 + hf * 40);
    uint4* dst = (uint4*)((unsigned int*)(smem + EOFF_W1) + rw * 84 + hf * 40);
#pragma unroll
    for (int j = 0; j < 10; ++j) dst[j] = src[j];
  }
  { // stage W2^T: 64 rows x 64 dwords -> LDS stride 68 dwords
    int rw = tid >> 2, q = tid & 3;
    const uint4* src = (const uint4*)(wt2 + rw * 64 + q * 16);
    uint4* dst = (uint4*)((unsigned int*)(smem + EOFF_W2) + rw * 68 + q * 16);
#pragma unroll
    for (int j = 0; j < 4; ++j) dst[j] = src[j];
  }
  { // gather edge inputs (bf16) into A-tile, LDS stride 84 dwords/row
    int e = tid & 127, which = tid >> 7;
    unsigned int* dstA = (unsigned int*)(smem + EOFF_A) + e * 84;
    if (which == 0) {
      int r = eidx[base + e];
      const uint4* src = (const uint4*)(xbf + (size_t)r * 32);
#pragma unroll
      for (int j = 0; j < 8; ++j) ((uint4*)dstA)[j] = src[j];
    } else {
      int c = eidx[N_EDGES + base + e];
      const uint4* src = (const uint4*)(xbf + (size_t)c * 32);
#pragma unroll
      for (int j = 0; j < 8; ++j) ((uint4*)(dstA + 32))[j] = src[j];
      const float4* ea4 = (const float4*)(eattr + (size_t)(base + e) * 32);
      uint4* d2 = (uint4*)(dstA + 64);
#pragma unroll
      for (int j = 0; j < 4; ++j) {
        float4 a = ea4[2 * j], b = ea4[2 * j + 1];
        d2[j] = uint4{pk2bf(a.x, a.y), pk2bf(a.z, a.w),
                      pk2bf(b.x, b.y), pk2bf(b.z, b.w)};
      }
    }
  }
  __syncthreads();

  const int lane = tid & 31, wv = tid >> 5;
  const int r = lane & 15, kh = lane >> 4;
  const uint4* aRow = (const uint4*)((unsigned int*)(smem + EOFF_A) + (16 * wv + r) * 84);

  // GEMM1: [16x160] @ [160x128], ReLU
  v8f acc1[8];
#pragma unroll
  for (int nt = 0; nt < 8; ++nt)
#pragma unroll
    for (int j = 0; j < 8; ++j) acc1[nt][j] = 0.f;
#pragma unroll
  for (int ks = 0; ks < 5; ++ks) {
    int k0 = ks * 32;
    v16bf af = loadA(aRow, k0, kh);
#pragma unroll
    for (int nt = 0; nt < 8; ++nt) {
      const uint4* bp = (const uint4*)((unsigned int*)(smem + EOFF_W1) + (nt * 16 + r) * 84);
      v16bf bf_ = loadB(bp, k0, kh);
      acc1[nt] = __builtin_amdgcn_wmma_f32_16x16x32_bf16(
          false, af, false, bf_, (short)0, acc1[nt], false, false);
    }
  }
  // bias + ReLU, write H (bf16) in place over this wave's own A rows
  unsigned short* sH = (unsigned short*)(smem + EOFF_A);
#pragma unroll
  for (int nt = 0; nt < 8; ++nt) {
    float b = sB1[nt * 16 + r];
#pragma unroll
    for (int j = 0; j < 8; ++j) {
      int m = 16 * wv + j + 8 * kh;
      float v = acc1[nt][j] + b;
      v = v > 0.f ? v : 0.f;
      sH[m * 168 + nt * 16 + r] = f2bf(v);
    }
  }

  // GEMM2: [16x128] @ [128x64]
  v8f acc2[4];
#pragma unroll
  for (int nt = 0; nt < 4; ++nt)
#pragma unroll
    for (int j = 0; j < 8; ++j) acc2[nt][j] = 0.f;
#pragma unroll
  for (int ks = 0; ks < 4; ++ks) {
    int k0 = ks * 32;
    v16bf af = loadA(aRow, k0, kh);
#pragma unroll
    for (int nt = 0; nt < 4; ++nt) {
      const uint4* bp = (const uint4*)((unsigned int*)(smem + EOFF_W2) + (nt * 16 + r) * 68);
      v16bf bf_ = loadB(bp, k0, kh);
      acc2[nt] = __builtin_amdgcn_wmma_f32_16x16x32_bf16(
          false, af, false, bf_, (short)0, acc2[nt], false, false);
    }
  }
  // store edge_embedding + scatter-add into aggregation by source node
#pragma unroll
  for (int nt = 0; nt < 4; ++nt) {
    float b = sB2[nt * 16 + r];
    int col = nt * 16 + r;
#pragma unroll
    for (int j = 0; j < 8; ++j) {
      int m = 16 * wv + j + 8 * kh;
      float v = acc2[nt][j] + b;
      eout[(size_t)(base + m) * 64 + col] = v;
      atomicAddF(&agg[(size_t)sRow[m] * 64 + col], v);
    }
  }
}

// ---------------- node MLP ----------------
__global__ void __launch_bounds__(256) node_kernel(
    const float* __restrict__ bn1, const float* __restrict__ bn2,
    const unsigned int* __restrict__ xbf,   // [N][32] dwords
    const unsigned int* __restrict__ wn1t,  // [128][64] dwords
    const unsigned int* __restrict__ wn2t,  // [64][64] dwords
    const float* __restrict__ agg,          // [N][64]
    float* __restrict__ nout)               // [N][64]
{
  extern __shared__ char smem[];
  float* sB1 = (float*)(smem + NOFF_B1);
  float* sB2 = (float*)(smem + NOFF_B2);

  const int tid  = threadIdx.x;
  const int base = blockIdx.x * 128;

  if (tid < 128) { sB1[tid] = bn1[tid]; if (tid < 64) sB2[tid] = bn2[tid]; }
  { // Wn1^T: 128 rows x 64 dwords -> LDS stride 68
    int rw = tid >> 1, hf = tid & 1;
    const uint4* src = (const uint4*)(wn1t + rw * 64 + hf * 32);
    uint4* dst = (uint4*)((unsigned int*)(smem + NOFF_W1) + rw * 68 + hf * 32);
#pragma unroll
    for (int j = 0; j < 8; ++j) dst[j] = src[j];
  }
  { // Wn2^T: 64 rows x 64 dwords -> LDS stride 68
    int rw = tid >> 2, q = tid & 3;
    const uint4* src = (const uint4*)(wn2t + rw * 64 + q * 16);
    uint4* dst = (uint4*)((unsigned int*)(smem + NOFF_W2) + rw * 68 + q * 16);
#pragma unroll
    for (int j = 0; j < 4; ++j) dst[j] = src[j];
  }
  { // A-tile: [x_bf16 | agg->bf16], LDS stride 68 dwords/row
    int nl = tid & 127, which = tid >> 7;
    int n = base + nl;
    unsigned int* dstA = (unsigned int*)(smem + NOFF_A) + nl * 68;
    if (which == 0) {
      if (n < N_NODES) {
        const uint4* src = (const uint4*)(xbf + (size_t)n * 32);
#pragma unroll
        for (int j = 0; j < 8; ++j) ((uint4*)dstA)[j] = src[j];
      } else {
#pragma unroll
        for (int j = 0; j < 8; ++j) ((uint4*)dstA)[j] = uint4{0u, 0u, 0u, 0u};
      }
    } else {
      uint4* d2 = (uint4*)(dstA + 32);
      if (n < N_NODES) {
        const float4* ag4 = (const float4*)(agg + (size_t)n * 64);
#pragma unroll
        for (int j = 0; j < 8; ++j) {
          float4 a = ag4[2 * j], b = ag4[2 * j + 1];
          d2[j] = uint4{pk2bf(a.x, a.y), pk2bf(a.z, a.w),
                        pk2bf(b.x, b.y), pk2bf(b.z, b.w)};
        }
      } else {
#pragma unroll
        for (int j = 0; j < 8; ++j) d2[j] = uint4{0u, 0u, 0u, 0u};
      }
    }
  }
  __syncthreads();

  const int lane = tid & 31, wv = tid >> 5;
  const int r = lane & 15, kh = lane >> 4;
  const uint4* aRow = (const uint4*)((unsigned int*)(smem + NOFF_A) + (16 * wv + r) * 68);

  v8f acc1[8];
#pragma unroll
  for (int nt = 0; nt < 8; ++nt)
#pragma unroll
    for (int j = 0; j < 8; ++j) acc1[nt][j] = 0.f;
#pragma unroll
  for (int ks = 0; ks < 4; ++ks) {
    int k0 = ks * 32;
    v16bf af = loadA(aRow, k0, kh);
#pragma unroll
    for (int nt = 0; nt < 8; ++nt) {
      const uint4* bp = (const uint4*)((unsigned int*)(smem + NOFF_W1) + (nt * 16 + r) * 68);
      v16bf bf_ = loadB(bp, k0, kh);
      acc1[nt] = __builtin_amdgcn_wmma_f32_16x16x32_bf16(
          false, af, false, bf_, (short)0, acc1[nt], false, false);
    }
  }
  unsigned short* sH = (unsigned short*)(smem + NOFF_A);
#pragma unroll
  for (int nt = 0; nt < 8; ++nt) {
    float b = sB1[nt * 16 + r];
#pragma unroll
    for (int j = 0; j < 8; ++j) {
      int m = 16 * wv + j + 8 * kh;
      float v = acc1[nt][j] + b;
      v = v > 0.f ? v : 0.f;
      sH[m * 136 + nt * 16 + r] = f2bf(v);
    }
  }

  v8f acc2[4];
#pragma unroll
  for (int nt = 0; nt < 4; ++nt)
#pragma unroll
    for (int j = 0; j < 8; ++j) acc2[nt][j] = 0.f;
#pragma unroll
  for (int ks = 0; ks < 4; ++ks) {
    int k0 = ks * 32;
    v16bf af = loadA(aRow, k0, kh);
#pragma unroll
    for (int nt = 0; nt < 4; ++nt) {
      const uint4* bp = (const uint4*)((unsigned int*)(smem + NOFF_W2) + (nt * 16 + r) * 68);
      v16bf bf_ = loadB(bp, k0, kh);
      acc2[nt] = __builtin_amdgcn_wmma_f32_16x16x32_bf16(
          false, af, false, bf_, (short)0, acc2[nt], false, false);
    }
  }
#pragma unroll
  for (int nt = 0; nt < 4; ++nt) {
    float b = sB2[nt * 16 + r];
    int col = nt * 16 + r;
#pragma unroll
    for (int j = 0; j < 8; ++j) {
      int m = 16 * wv + j + 8 * kh;
      int n = base + m;
      if (n < N_NODES) nout[(size_t)n * 64 + col] = acc2[nt][j] + b;
    }
  }
}

extern "C" void kernel_launch(void* const* d_in, const int* in_sizes, int n_in,
                              void* d_out, int out_size, void* d_ws, size_t ws_size,
                              hipStream_t stream) {
  const float* x     = (const float*)d_in[0];
  const int*   eidx  = (const int*)  d_in[1];
  const float* eattr = (const float*)d_in[2];
  const float* We1   = (const float*)d_in[3];
  const float* be1   = (const float*)d_in[4];
  const float* We2   = (const float*)d_in[5];
  const float* be2   = (const float*)d_in[6];
  const float* Wn1   = (const float*)d_in[7];
  const float* bn1   = (const float*)d_in[8];
  const float* Wn2   = (const float*)d_in[9];
  const float* bn2   = (const float*)d_in[10];

  char* ws = (char*)d_ws;
  float*        agg  = (float*)       (ws + WS_AGG);
  unsigned int* xbf  = (unsigned int*)(ws + WS_XBF);
  unsigned int* wt1  = (unsigned int*)(ws + WS_WT1);
  unsigned int* wt2  = (unsigned int*)(ws + WS_WT2);
  unsigned int* wn1t = (unsigned int*)(ws + WS_WN1T);
  unsigned int* wn2t = (unsigned int*)(ws + WS_WN2T);

  float* eout = (float*)d_out;                      // [E][64]
  float* nout = eout + (size_t)N_EDGES * 64;        // [N][64]

  zero_agg<<<(N_NODES * 64) / (256 * 4), 256, 0, stream>>>((float4*)agg);
  prep_convert<<<2048, 256, 0, stream>>>(x, We1, We2, Wn1, Wn2,
                                         xbf, wt1, wt2, wn1t, wn2t);
  edge_kernel<<<N_EDGES / 128, 256, ESMEM, stream>>>(
      eidx, eattr, be1, be2, xbf, wt1, wt2, agg, eout);
  node_kernel<<<(N_NODES + 127) / 128, 256, NSMEM, stream>>>(
      bn1, bn2, xbf, wn1t, wn2t, agg, nout);
}
